// MambaBlock_49486613184658
// MI455X (gfx1250) — matrix operand order repaired and checked by reading, not compiled, not run on hardware
//
#include <hip/hip_runtime.h>
#include <hip/hip_bf16.h>
#include <math.h>

// ---- problem constants (match reference) ----
#define Bc     2
#define Lc     2048
#define Dc     1024
#define DIc    2048
#define NSTATE 16
#define DCONVc 4
#define DTRc   64
#define Ic     4096
#define MROWS  (Bc * Lc)   // 4096

// ---- CDNA5 WMMA types ----
typedef __bf16 bf16;
typedef bf16  v16bf __attribute__((ext_vector_type(16)));
typedef bf16  v8bf  __attribute__((ext_vector_type(8)));
typedef float v8f   __attribute__((ext_vector_type(8)));

__device__ __forceinline__ bf16 f2bf(float f) {
  unsigned u = __builtin_bit_cast(unsigned, f);
  unsigned r = u + 0x7FFFu + ((u >> 16) & 1u);   // round-to-nearest-even
  unsigned short h = (unsigned short)(r >> 16);
  return __builtin_bit_cast(bf16, h);
}

__device__ __forceinline__ v16bf cat8(v8bf lo, v8bf hi) {
  v16bf r;
#pragma unroll
  for (int i = 0; i < 8; ++i) { r[i] = lo[i]; r[i + 8] = hi[i]; }
  return r;
}

__device__ __forceinline__ float sigmoidf_(float x) { return 1.f / (1.f + __expf(-x)); }

// ---------------------------------------------------------------------------
// fp32 -> bf16 conversion (weights / activations)
// ---------------------------------------------------------------------------
__global__ void k_cvt_bf16(const float* __restrict__ in, bf16* __restrict__ out, int n) {
  int i = blockIdx.x * blockDim.x + threadIdx.x;
  if (i < n) out[i] = f2bf(in[i]);
}

// ---------------------------------------------------------------------------
// Fused (optional add) + LayerNorm. One block (256 thr) per row of D=1024.
// Writes res_out (f32, if TWO) and normalized output as bf16.
// ---------------------------------------------------------------------------
template <bool TWO>
__global__ __launch_bounds__(256)
void k_ln(const float* __restrict__ a, const float* __restrict__ b,
          const float* __restrict__ w, const float* __restrict__ bias,
          float* __restrict__ res_out, bf16* __restrict__ hbf) {
  __shared__ float s1[256];
  __shared__ float s2[256];
  const int row = blockIdx.x;
  const int tid = threadIdx.x;
  float4 x = ((const float4*)(a + (size_t)row * Dc))[tid];
  if (TWO) {
    float4 y = ((const float4*)(b + (size_t)row * Dc))[tid];
    x.x += y.x; x.y += y.y; x.z += y.z; x.w += y.w;
    ((float4*)(res_out + (size_t)row * Dc))[tid] = x;
  }
  s1[tid] = x.x + x.y + x.z + x.w;
  s2[tid] = x.x * x.x + x.y * x.y + x.z * x.z + x.w * x.w;
  __syncthreads();
  for (int st = 128; st > 0; st >>= 1) {
    if (tid < st) { s1[tid] += s1[tid + st]; s2[tid] += s2[tid + st]; }
    __syncthreads();
  }
  const float mean = s1[0] * (1.f / Dc);
  const float var  = s2[0] * (1.f / Dc) - mean * mean;
  const float inv  = rsqrtf(var + 1e-5f);
  bf16* ho = hbf + (size_t)row * Dc;
  float v[4] = {x.x, x.y, x.z, x.w};
#pragma unroll
  for (int i = 0; i < 4; ++i) {
    int c = tid * 4 + i;
    ho[c] = f2bf((v[i] - mean) * inv * w[c] + bias[c]);
  }
}

// ---------------------------------------------------------------------------
// WMMA bf16 GEMM:  C[M,N] = A[M,K] * W[N,K]^T   (einsum 'mk,nk->mn')
// One wave computes a (16*MI) x (16*NJ) tile.  Per K-step: MI A-frags and
// NJ B-frags are ALL loaded first (one load batch in flight), then MI*NJ
// back-to-back WMMAs.  Grid covers M,N exactly -> straight-line K-loop.
// VGPR layouts per CDNA5 ISA 7.12.2 (16-bit A 16x32; B columns per lane).
// ---------------------------------------------------------------------------
enum { EPI_NONE = 0, EPI_SOFTPLUS = 1, EPI_ADD = 2, EPI_SILU_MUL_BF16 = 3 };

template <int EPI, int NJ, int MI>
__global__ __launch_bounds__(32)
void k_wmma_gemm(const bf16* __restrict__ A, int lda,
                 const bf16* __restrict__ W, int ldw,
                 float* __restrict__ C, int ldc,
                 bf16* __restrict__ Cbf,
                 const float* __restrict__ extra, int ldx,
                 int K) {
  const int lane = threadIdx.x;          // wave32
  const int half = lane >> 4;            // 0 | 1
  const int lm   = lane & 15;
  const int m0   = blockIdx.x * (16 * MI);
  const int n0b  = blockIdx.y * (16 * NJ);

  v8f acc[MI][NJ] = {};

  // A: lanes 0-15 -> row m, K {k0..k0+7, k0+16..k0+23}; lanes 16-31 -> {+8, +24}
  const bf16* arow[MI];
#pragma unroll
  for (int i = 0; i < MI; ++i)
    arow[i] = A + (size_t)(m0 + i * 16 + lm) * lda + half * 8;
  // B: lane n (0-15) holds column n0+n, K {k0..k0+7, k0+16..k0+23}; lanes 16-31 -> {+8, +24}
  const bf16* wrow[NJ];
#pragma unroll
  for (int j = 0; j < NJ; ++j)
    wrow[j] = W + (size_t)(n0b + j * 16 + lm) * ldw + half * 8;

  for (int k0 = 0; k0 < K; k0 += 32) {
    v16bf af[MI];
    v16bf bfr[NJ];
#pragma unroll
    for (int i = 0; i < MI; ++i)
      af[i] = cat8(*(const v8bf*)(arow[i] + k0),
                   *(const v8bf*)(arow[i] + k0 + 16));
#pragma unroll
    for (int j = 0; j < NJ; ++j)
      bfr[j] = cat8(*(const v8bf*)(wrow[j] + k0),
                    *(const v8bf*)(wrow[j] + k0 + 16));
    __builtin_prefetch(arow[0] + k0 + 32, 0, 3);   // global_prefetch_b8 (near)
#pragma unroll
    for (int i = 0; i < MI; ++i)
#pragma unroll
      for (int j = 0; j < NJ; ++j)
        acc[i][j] = __builtin_amdgcn_wmma_f32_16x16x32_bf16(
            false, af[i], false, bfr[j], (short)0, acc[i][j], false, false);
  }

  // C/D layout: lanes 0-15 -> M = +0..7 (vgpr r), N = lane; lanes 16-31 -> M = +8..15
#pragma unroll
  for (int i = 0; i < MI; ++i) {
#pragma unroll
    for (int j = 0; j < NJ; ++j) {
      const int col = n0b + j * 16 + lm;
#pragma unroll
      for (int r = 0; r < 8; ++r) {
        const int row = m0 + i * 16 + half * 8 + r;
        float v = acc[i][j][r];
        const size_t ci = (size_t)row * ldc + col;
        if (EPI == EPI_NONE) {
          C[ci] = v;
        } else if (EPI == EPI_SOFTPLUS) {            // softplus(acc + bias[col])
          float t = v + extra[col];
          C[ci] = (t > 20.f) ? t : log1pf(__expf(t));
        } else if (EPI == EPI_ADD) {                 // acc + addend[row,col]
          C[ci] = v + extra[(size_t)row * ldx + col];
        } else {                                     // bf16( silu(acc) * mul[row,col] )
          float u = extra[(size_t)row * ldx + col];
          Cbf[ci] = f2bf(v * sigmoidf_(v) * u);
        }
      }
    }
  }
}

// ---------------------------------------------------------------------------
// Causal depthwise conv1d (k=4) + SiLU, with direction-aware time mapping.
// Reads x-half of xz (f32), writes f32 + bf16 in scan-time order.
// ---------------------------------------------------------------------------
__global__ __launch_bounds__(256)
void k_conv(const float* __restrict__ xz, const float* __restrict__ cw,
            const float* __restrict__ cb, float* __restrict__ xo,
            bf16* __restrict__ xobf, int dir) {
  const int rt = blockIdx.x;             // b*L + t  (scan time)
  const int b = rt / Lc, t = rt % Lc;
  for (int c = threadIdx.x; c < DIc; c += 256) {
    float acc = cb[c];
#pragma unroll
    for (int k = 0; k < DCONVc; ++k) {
      int tt = t - (DCONVc - 1) + k;
      if (tt >= 0) {
        int sl = dir ? (Lc - 1 - tt) : tt;
        acc += cw[c * DCONVc + k] * xz[((size_t)(b * Lc + sl)) * (2 * DIc) + c];
      }
    }
    float s = acc * sigmoidf_(acc);      // silu
    const size_t o = (size_t)rt * DIc + c;
    xo[o] = s;
    xobf[o] = f2bf(s);
  }
}

// ---------------------------------------------------------------------------
// Selective scan (S6). One thread per channel c, 16 states in registers.
// B/C (per-timestep, channel-shared) staged through LDS each step.
// Epilogue fuses  y = (scan + x*D) * silu(z), with direction-mapped z.
// ---------------------------------------------------------------------------
__global__ __launch_bounds__(256)
void k_scan(const float* __restrict__ xconv, const float* __restrict__ dtv,
            const float* __restrict__ xdbl, const float* __restrict__ xz,
            const float* __restrict__ A_log, const float* __restrict__ Dp,
            float* __restrict__ yout, int dir) {
  __shared__ float sB[NSTATE];
  __shared__ float sC[NSTATE];
  const int bpb = DIc / 256;             // blocks per (b)
  const int b = blockIdx.x / bpb;
  const int c = (blockIdx.x % bpb) * 256 + threadIdx.x;

  float s[NSTATE], Arow[NSTATE];
#pragma unroll
  for (int n = 0; n < NSTATE; ++n) {
    s[n] = 0.f;
    Arow[n] = -__expf(A_log[(size_t)c * NSTATE + n]);
  }
  const float Dv = Dp[c];

  for (int t = 0; t < Lc; ++t) {
    const int rt = b * Lc + t;
    if (threadIdx.x < 2 * NSTATE) {
      float v = xdbl[(size_t)rt * (DTRc + 2 * NSTATE) + DTRc + threadIdx.x];
      if (threadIdx.x < NSTATE) sB[threadIdx.x] = v;
      else                      sC[threadIdx.x - NSTATE] = v;
    }
    __syncthreads();
    const float dt  = dtv[(size_t)rt * DIc + c];
    const float xv  = xconv[(size_t)rt * DIc + c];
    const float dtx = dt * xv;
    float y = 0.f;
#pragma unroll
    for (int n = 0; n < NSTATE; ++n) {
      s[n] = s[n] * __expf(dt * Arow[n]) + dtx * sB[n];
      y += s[n] * sC[n];
    }
    const int zl = dir ? (Lc - 1 - t) : t;
    const float zv = xz[((size_t)(b * Lc + zl)) * (2 * DIc) + DIc + c];
    yout[(size_t)rt * DIc + c] = (y + xv * Dv) * (zv * sigmoidf_(zv));
    __syncthreads();
  }
}

// ---------------------------------------------------------------------------
// y_sum = y_f[t] + y_r[L-1-t]  ->  bf16 (input to shared outproj GEMM)
// ---------------------------------------------------------------------------
__global__ void k_combine(const float* __restrict__ yf, const float* __restrict__ yr,
                          bf16* __restrict__ out) {
  size_t i = (size_t)blockIdx.x * blockDim.x + threadIdx.x;
  if (i >= (size_t)Bc * Lc * DIc) return;
  int c = (int)(i % DIc);
  size_t rt = i / DIc;
  int b = (int)(rt / Lc), t = (int)(rt % Lc);
  float v = yf[i] + yr[((size_t)(b * Lc + (Lc - 1 - t))) * DIc + c];
  out[i] = f2bf(v);
}

// ---------------------------------------------------------------------------
// host side
// ---------------------------------------------------------------------------
static void gemm(hipStream_t s, int epi, const bf16* A, int lda, const bf16* W, int ldw,
                 float* C, int ldc, bf16* Cbf, const float* extra, int ldx, int N, int K) {
  dim3 b(32);
  if (N % 64 == 0) {
    dim3 g(MROWS / 32, N / 64);          // MI=2, NJ=4
    switch (epi) {
      case EPI_NONE:     k_wmma_gemm<EPI_NONE, 4, 2><<<g, b, 0, s>>>(A, lda, W, ldw, C, ldc, Cbf, extra, ldx, K); break;
      case EPI_SOFTPLUS: k_wmma_gemm<EPI_SOFTPLUS, 4, 2><<<g, b, 0, s>>>(A, lda, W, ldw, C, ldc, Cbf, extra, ldx, K); break;
      case EPI_ADD:      k_wmma_gemm<EPI_ADD, 4, 2><<<g, b, 0, s>>>(A, lda, W, ldw, C, ldc, Cbf, extra, ldx, K); break;
      default:           k_wmma_gemm<EPI_SILU_MUL_BF16, 4, 2><<<g, b, 0, s>>>(A, lda, W, ldw, C, ldc, Cbf, extra, ldx, K); break;
    }
  } else {               // N % 32 == 0 (xproj: N = 96)
    dim3 g(MROWS / 32, N / 32);          // MI=2, NJ=2
    k_wmma_gemm<EPI_NONE, 2, 2><<<g, b, 0, s>>>(A, lda, W, ldw, C, ldc, Cbf, extra, ldx, K);
  }
}

static void cvt(hipStream_t s, const float* in, bf16* out, size_t n) {
  k_cvt_bf16<<<dim3((unsigned)((n + 255) / 256)), dim3(256), 0, s>>>(in, out, (int)n);
}

extern "C" void kernel_launch(void* const* d_in, const int* in_sizes, int n_in,
                              void* d_out, int out_size, void* d_ws, size_t ws_size,
                              hipStream_t stream) {
  (void)in_sizes; (void)n_in; (void)out_size; (void)ws_size;

  const float* hidden   = (const float*)d_in[0];
  const float* residual = (const float*)d_in[1];
  const float* inproj_w = (const float*)d_in[2];
  const float* outproj_w= (const float*)d_in[3];
  const float* conv_w[2]  = {(const float*)d_in[4],  (const float*)d_in[11]};
  const float* conv_b[2]  = {(const float*)d_in[5],  (const float*)d_in[12]};
  const float* xproj_w[2] = {(const float*)d_in[6],  (const float*)d_in[13]};
  const float* dtproj_w[2]= {(const float*)d_in[7],  (const float*)d_in[14]};
  const float* dtproj_b[2]= {(const float*)d_in[8],  (const float*)d_in[15]};
  const float* A_log[2]   = {(const float*)d_in[9],  (const float*)d_in[16]};
  const float* Dp[2]      = {(const float*)d_in[10], (const float*)d_in[17]};
  const float* ln1_w = (const float*)d_in[18];
  const float* ln1_b = (const float*)d_in[19];
  const float* ln2_w = (const float*)d_in[20];
  const float* ln2_b = (const float*)d_in[21];
  const float* gate_w = (const float*)d_in[22];
  const float* up_w   = (const float*)d_in[23];
  const float* down_w = (const float*)d_in[24];

  float* out_h    = (float*)d_out;                        // (B,L,D)
  float* resid2   = (float*)d_out + (size_t)MROWS * Dc;   // (B,L,D)

  // ---- workspace layout ----
  char* ws = (char*)d_ws;
  size_t off = 0;
  auto alloc = [&](size_t bytes) -> void* {
    void* p = ws + off;
    off = (off + bytes + 255) & ~(size_t)255;
    return p;
  };
  bf16* bf_inproj  = (bf16*)alloc((size_t)2 * DIc * Dc * 2);
  bf16* bf_outproj = (bf16*)alloc((size_t)Dc * DIc * 2);
  bf16* bf_xproj[2]  = {(bf16*)alloc((size_t)(DTRc + 2 * NSTATE) * DIc * 2),
                        (bf16*)alloc((size_t)(DTRc + 2 * NSTATE) * DIc * 2)};
  bf16* bf_dtproj[2] = {(bf16*)alloc((size_t)DIc * DTRc * 2),
                        (bf16*)alloc((size_t)DIc * DTRc * 2)};
  bf16* bf_gate = (bf16*)alloc((size_t)Ic * Dc * 2);
  bf16* bf_up   = (bf16*)alloc((size_t)Ic * Dc * 2);
  bf16* bf_down = (bf16*)alloc((size_t)Dc * Ic * 2);

  bf16*  h_bf  = (bf16*)alloc((size_t)MROWS * Dc * 2);
  float* res1  = (float*)alloc((size_t)MROWS * Dc * 4);
  float* xz    = (float*)alloc((size_t)MROWS * 2 * DIc * 4);
  float* xconv[2]   = {(float*)alloc((size_t)MROWS * DIc * 4), (float*)alloc((size_t)MROWS * DIc * 4)};
  bf16*  xconvbf[2] = {(bf16*)alloc((size_t)MROWS * DIc * 2),  (bf16*)alloc((size_t)MROWS * DIc * 2)};
  float* xdbl[2]    = {(float*)alloc((size_t)MROWS * 96 * 4),  (float*)alloc((size_t)MROWS * 96 * 4)};
  bf16*  xdblbf[2]  = {(bf16*)alloc((size_t)MROWS * 96 * 2),   (bf16*)alloc((size_t)MROWS * 96 * 2)};
  float* dtf[2]     = {(float*)alloc((size_t)MROWS * DIc * 4), (float*)alloc((size_t)MROWS * DIc * 4)};
  float* yv[2]      = {(float*)alloc((size_t)MROWS * DIc * 4), (float*)alloc((size_t)MROWS * DIc * 4)};
  bf16*  ysum_bf = (bf16*)alloc((size_t)MROWS * DIc * 2);
  bf16*  h2_bf   = (bf16*)alloc((size_t)MROWS * Dc * 2);
  float* up_f    = (float*)alloc((size_t)MROWS * Ic * 4);
  bf16*  gu_bf   = (bf16*)alloc((size_t)MROWS * Ic * 2);

  // ---- weight conversion fp32 -> bf16 ----
  cvt(stream, inproj_w,  bf_inproj,  (size_t)2 * DIc * Dc);
  cvt(stream, outproj_w, bf_outproj, (size_t)Dc * DIc);
  for (int d = 0; d < 2; ++d) {
    cvt(stream, xproj_w[d],  bf_xproj[d],  (size_t)(DTRc + 2 * NSTATE) * DIc);
    cvt(stream, dtproj_w[d], bf_dtproj[d], (size_t)DIc * DTRc);
  }
  cvt(stream, gate_w, bf_gate, (size_t)Ic * Dc);
  cvt(stream, up_w,   bf_up,   (size_t)Ic * Dc);
  cvt(stream, down_w, bf_down, (size_t)Dc * Ic);

  // 1) res1 = hidden + residual ; h = LN1(res1)
  k_ln<true><<<MROWS, 256, 0, stream>>>(hidden, residual, ln1_w, ln1_b, res1, h_bf);

  // 2) xz = h @ inproj^T   (shared between directions; reversal is a row permutation)
  gemm(stream, EPI_NONE, h_bf, Dc, bf_inproj, Dc, xz, 2 * DIc, nullptr, nullptr, 0, 2 * DIc, Dc);

  for (int d = 0; d < 2; ++d) {
    // 3) causal depthwise conv + silu (direction-mapped time)
    k_conv<<<MROWS, 256, 0, stream>>>(xz, conv_w[d], conv_b[d], xconv[d], xconvbf[d], d);
    // 4) x_dbl = xconv @ xproj^T  (N = 96 -> NJ=2 path)
    gemm(stream, EPI_NONE, xconvbf[d], DIc, bf_xproj[d], DIc, xdbl[d], 96, nullptr, nullptr, 0, 96, DIc);
    cvt(stream, xdbl[d], xdblbf[d], (size_t)MROWS * 96);
    // 5) dt = softplus(x_dbl[:, :64] @ dtproj^T + b)
    gemm(stream, EPI_SOFTPLUS, xdblbf[d], 96, bf_dtproj[d], DTRc, dtf[d], DIc, nullptr, dtproj_b[d], 0, DIc, DTRc);
    // 6) selective scan + fused (y + x*D) * silu(z)
    k_scan<<<Bc * (DIc / 256), 256, 0, stream>>>(xconv[d], dtf[d], xdbl[d], xz,
                                                 A_log[d], Dp[d], yv[d], d);
  }

  // 7) y_sum = y_f + reverse(y_r) -> bf16 ; outproj once with +res1 epilogue -> residual2
  {
    size_t n = (size_t)MROWS * DIc;
    k_combine<<<(unsigned)((n + 255) / 256), 256, 0, stream>>>(yv[0], yv[1], ysum_bf);
  }
  gemm(stream, EPI_ADD, ysum_bf, DIc, bf_outproj, DIc, resid2, Dc, nullptr, res1, Dc, Dc, DIc);

  // 8) h2 = LN2(residual2)
  k_ln<false><<<MROWS, 256, 0, stream>>>(resid2, nullptr, ln2_w, ln2_b, nullptr, h2_bf);

  // 9) MLP: up, gate (fused silu*up -> bf16), down -> out_h
  gemm(stream, EPI_NONE, h2_bf, Dc, bf_up, Dc, up_f, Ic, nullptr, nullptr, 0, Ic, Dc);
  gemm(stream, EPI_SILU_MUL_BF16, h2_bf, Dc, bf_gate, Dc, nullptr, Ic, gu_bf, up_f, Ic, Ic, Dc);
  gemm(stream, EPI_NONE, gu_bf, Ic, bf_down, Ic, out_h, Dc, nullptr, nullptr, 0, Dc, Ic);
}
